// MultiHeadAttention_80418967650540
// MI455X (gfx1250) — compile-verified, matrix operand order checked by reading, not run on hardware
//
#include <hip/hip_runtime.h>
#include <hip/hip_bf16.h>

// ---------------------------------------------------------------------------
// MI455X (gfx1250) multi-head attention forward.
// Memory-bound on the 537MB attn output -> bf16 WMMA everywhere, f32 accum,
// single round-trip of the score buffer (d_out attn region used as scratch).
// GEMMs stage shared B tiles in LDS via async global->LDS (double buffered).
// ---------------------------------------------------------------------------

typedef __bf16 bf16_t;
typedef __bf16  v8bf  __attribute__((ext_vector_type(8)));
typedef __bf16  v16bf __attribute__((ext_vector_type(16)));
typedef float   v8f   __attribute__((ext_vector_type(8)));

#define D_MODEL 1024
#define SEQ     2048
#define NBATCH  2
#define NHEAD   16
#define DK      64
#define BH      (NBATCH * NHEAD)       // 32
#define MTOT    (NBATCH * SEQ)         // 4096

#if defined(__has_builtin)
#if __has_builtin(__builtin_amdgcn_global_load_async_to_lds_b128) && \
    __has_builtin(__builtin_amdgcn_s_wait_asynccnt)
#define ASYNC_LDS 1
#endif
#endif
#ifndef ASYNC_LDS
#define ASYNC_LDS 0
#endif

#if ASYNC_LDS
// Builtin signature (from hipcc diagnostic): param0 is
//   'int __attribute__((vector_size(16))) __device__ *'  (AS1),
// param1 is the LDS-side pointer (AS3).
typedef int v4i_vs __attribute__((vector_size(16)));
typedef __attribute__((address_space(1))) v4i_vs as1_v4i;
typedef __attribute__((address_space(3))) v4i_vs as3_v4i;
#endif

// ---------------------------------------------------------------------------
// WMMA helper: D = A(16x32 bf16) * B(32x16 bf16) + C(16x16 f32)
// ---------------------------------------------------------------------------
__device__ __forceinline__ v8f wmma_bf16(v16bf a, v16bf b, v8f c) {
    return __builtin_amdgcn_wmma_f32_16x16x32_bf16(
        /*neg_a=*/false, a, /*neg_b=*/false, b,
        /*c_mod=*/(short)0, c, /*reuse_a=*/false, /*reuse_b=*/false);
}

// Load one 16-lane fragment slice per ISA 7.12.2 layout:
// lane holds K = {sel..sel+7} at p[0..7] and K = {sel+16..sel+23} at p[8..15].
__device__ __forceinline__ v16bf frag_from(const bf16_t* __restrict__ p) {
    v8bf lo = *(const v8bf*)(p);
    v8bf hi = *(const v8bf*)(p + 16);
    v16bf r;
#pragma unroll
    for (int i = 0; i < 8; ++i) { r[i] = lo[i]; r[i + 8] = hi[i]; }
    return r;
}

// ---------------------------------------------------------------------------
// LDS staging of a 64(col) x 32(k) bf16 B tile, one 16B chunk per thread.
// Row stride 48 elements (96B): keeps every chunk 16B-aligned for b128 ops
// and spreads LDS banks.
// ---------------------------------------------------------------------------
#define BSTRIDE 48
#define BTILE_ELEMS (64 * BSTRIDE)

__device__ __forceinline__ void stage_b_tile(const bf16_t* __restrict__ wT,
                                             int n0, int k,
                                             bf16_t* __restrict__ buf) {
    const int t   = threadIdx.x;
    const int col = t >> 2;            // 0..63
    const int kc  = (t & 3) * 8;       // 0,8,16,24
    const bf16_t* src = wT + (size_t)(n0 + col) * D_MODEL + k + kc;
    bf16_t* dst = buf + col * BSTRIDE + kc;
#if ASYNC_LDS
    __builtin_amdgcn_global_load_async_to_lds_b128(
        (as1_v4i*)src, (as3_v4i*)dst, 0, 0);
#else
    *(v8bf*)dst = *(const v8bf*)src;
#endif
}

__device__ __forceinline__ void stage_commit() {
#if ASYNC_LDS
    __builtin_amdgcn_s_wait_asynccnt(0);
#endif
    __syncthreads();
}

// ---------------------------------------------------------------------------
// Elementwise f32 -> bf16 convert
// ---------------------------------------------------------------------------
__global__ void __launch_bounds__(256) cvt_bf16_kernel(const float* __restrict__ in,
                                                       bf16_t* __restrict__ out, int n) {
    int i = blockIdx.x * 256 + threadIdx.x;
    if (i < n) out[i] = (bf16_t)in[i];
}

// in: [R, C] f32 row-major.  out: [C, R] bf16 (i.e. in^T), so that WMMA
// B-operand fragments become contiguous 16B loads.
__global__ void __launch_bounds__(256) transpose_cvt_kernel(const float* __restrict__ in,
                                                            bf16_t* __restrict__ out,
                                                            int R, int C) {
    int i = blockIdx.x * 256 + threadIdx.x;
    if (i >= R * C) return;
    int r = i / C, c = i - r * C;
    out[(size_t)c * R + r] = (bf16_t)in[i];
}

// ---------------------------------------------------------------------------
// QKV projection: qkv[m, n] = sum_k xb[m,k] * Wqkv[k,n] + bias[n]
// A = xb [4096,1024] bf16 rm, B^T = wT [3072,1024] bf16.
// Block = 8 waves (128x64 tile); B tile double-buffered through LDS.
// Epilogue scatters q->[BH,L,DK], k->[BH,L,DK], v->[BH,DK,L] (transposed).
// ---------------------------------------------------------------------------
__global__ void __launch_bounds__(256) gemm_qkv_kernel(const bf16_t* __restrict__ xb,
                                                       const bf16_t* __restrict__ wT,
                                                       const float* __restrict__ bias,
                                                       bf16_t* __restrict__ qb,
                                                       bf16_t* __restrict__ kb,
                                                       bf16_t* __restrict__ vt) {
    __shared__ bf16_t ldsB[2][BTILE_ELEMS];
    const int lane = threadIdx.x & 31;
    const int wave = threadIdx.x >> 5;
    const int m0   = blockIdx.x * 128 + wave * 16;
    const int n0   = blockIdx.y * 64;
    const int lr   = lane & 15;
    const int sel  = (lane >> 4) * 8;

    v8f c[4] = {};
    const bf16_t* arow = xb + (size_t)(m0 + lr) * D_MODEL;

    stage_b_tile(wT, n0, 0, ldsB[0]);
    stage_commit();
#pragma unroll 2
    for (int kt = 0; kt < D_MODEL / 32; ++kt) {
        const int cur = kt & 1;
        if (kt + 1 < D_MODEL / 32)
            stage_b_tile(wT, n0, (kt + 1) * 32, ldsB[cur ^ 1]);
        v16bf a = frag_from(arow + kt * 32 + sel);
#pragma unroll
        for (int nt = 0; nt < 4; ++nt) {
            v16bf b = frag_from(&ldsB[cur][(nt * 16 + lr) * BSTRIDE + sel]);
            c[nt] = wmma_bf16(a, b, c[nt]);
        }
        stage_commit();
    }

#pragma unroll
    for (int nt = 0; nt < 4; ++nt) {
        const int n   = n0 + nt * 16 + lr;
        const float bv = bias[n];
        const int mat = n >> 10;        // 0=q 1=k 2=v
        const int nn  = n & 1023;
        const int h   = nn >> 6;
        const int dk  = nn & 63;
#pragma unroll
        for (int v = 0; v < 8; ++v) {
            const int m  = m0 + v + ((lane >> 4) << 3);
            const int b_ = m >> 11;                 // batch
            const int l  = m & (SEQ - 1);
            const bf16_t val = (bf16_t)(c[nt][v] + bv);
            const int bh = b_ * NHEAD + h;
            if (mat == 0)
                qb[((size_t)bh * SEQ + l) * DK + dk] = val;
            else if (mat == 1)
                kb[((size_t)bh * SEQ + l) * DK + dk] = val;
            else
                vt[((size_t)bh * DK + dk) * SEQ + l] = val;
        }
    }
}

// ---------------------------------------------------------------------------
// Scores pass: S = (q @ k^T) / 8, streamed per 16x16 key tile.
// Writes scaled raw scores into the attn region of d_out (scratch), keeps
// online (rowmax, rowsumexp) per query row -> stats[BH*L][2].
// wave = one 16-row query tile. grid (32, 16) x 8 waves.
// ---------------------------------------------------------------------------
__global__ void __launch_bounds__(256) attn_scores_kernel(const bf16_t* __restrict__ qb,
                                                          const bf16_t* __restrict__ kb,
                                                          float* __restrict__ sc,
                                                          float* __restrict__ stats) {
    const int lane = threadIdx.x & 31;
    const int wave = threadIdx.x >> 5;
    const int bh   = blockIdx.x;
    const int q0   = (blockIdx.y * 8 + wave) * 16;
    const int lr   = lane & 15;
    const int sel  = (lane >> 4) * 8;
    const int hi8  = (lane >> 4) << 3;

    const bf16_t* qrow = qb + ((size_t)bh * SEQ + q0 + lr) * DK;
    const v16bf a0 = frag_from(qrow + sel);
    const v16bf a1 = frag_from(qrow + 32 + sel);

    float mrow[8], srow[8];
#pragma unroll
    for (int i = 0; i < 8; ++i) { mrow[i] = -__builtin_inff(); srow[i] = 0.0f; }

    const float scale = 0.125f;  // 1/sqrt(64)
    float* scbase = sc + ((size_t)bh * SEQ + q0) * SEQ;

    for (int kt = 0; kt < SEQ / 16; ++kt) {
        const bf16_t* krow = kb + ((size_t)bh * SEQ + kt * 16 + lr) * DK;
        __builtin_prefetch(krow + 16 * DK, 0, 0);   // next key tile
        v16bf b0 = frag_from(krow + sel);
        v16bf b1 = frag_from(krow + 32 + sel);
        v8f c = {};
        c = wmma_bf16(a0, b0, c);
        c = wmma_bf16(a1, b1, c);
#pragma unroll
        for (int v = 0; v < 8; ++v) {
            const float s = c[v] * scale;
            scbase[(size_t)(v + hi8) * SEQ + kt * 16 + lr] = s;
            // half-wave (16-lane) row reductions
            float mx = s;
            mx = fmaxf(mx, __shfl_xor(mx, 1));
            mx = fmaxf(mx, __shfl_xor(mx, 2));
            mx = fmaxf(mx, __shfl_xor(mx, 4));
            mx = fmaxf(mx, __shfl_xor(mx, 8));
            const float mn = fmaxf(mrow[v], mx);
            float p = __expf(s - mn);
            p += __shfl_xor(p, 1);
            p += __shfl_xor(p, 2);
            p += __shfl_xor(p, 4);
            p += __shfl_xor(p, 8);
            srow[v] = srow[v] * __expf(mrow[v] - mn) + p;
            mrow[v] = mn;
        }
    }
    if (lr == 0) {
#pragma unroll
        for (int v = 0; v < 8; ++v) {
            const size_t idx = (size_t)bh * SEQ + q0 + hi8 + v;
            stats[idx * 2 + 0] = mrow[v];
            stats[idx * 2 + 1] = srow[v];
        }
    }
}

// ---------------------------------------------------------------------------
// Ctx pass: attn = exp(S - m)/sum written in place (final output), and
// ctx(16x64) = attn @ v via WMMA with vt ([BH,DK,L] so B frags are contiguous).
// ---------------------------------------------------------------------------
__global__ void __launch_bounds__(256) attn_ctx_kernel(const bf16_t* __restrict__ vt,
                                                       const float* __restrict__ stats,
                                                       float* __restrict__ attn,
                                                       bf16_t* __restrict__ ctxb) {
    const int lane = threadIdx.x & 31;
    const int wave = threadIdx.x >> 5;
    const int bh   = blockIdx.x;
    const int q0   = (blockIdx.y * 8 + wave) * 16;
    const int lr   = lane & 15;
    const int sel  = (lane >> 4) * 8;

    const float2 st = ((const float2*)stats)[(size_t)bh * SEQ + q0 + lr];
    const float mrow = st.x;
    const float inv  = 1.0f / st.y;

    float* arow = attn + ((size_t)bh * SEQ + q0 + lr) * SEQ;
    v8f c[4] = {};

    for (int ks = 0; ks < SEQ; ks += 32) {
        float4 r0 = *(const float4*)(arow + ks + sel);
        float4 r1 = *(const float4*)(arow + ks + sel + 4);
        float4 r2 = *(const float4*)(arow + ks + sel + 16);
        float4 r3 = *(const float4*)(arow + ks + sel + 20);
        __builtin_prefetch(arow + ks + 32 + sel, 0, 0);   // next score chunk
        float p[16];
        p[0] = r0.x; p[1] = r0.y; p[2]  = r0.z; p[3]  = r0.w;
        p[4] = r1.x; p[5] = r1.y; p[6]  = r1.z; p[7]  = r1.w;
        p[8] = r2.x; p[9] = r2.y; p[10] = r2.z; p[11] = r2.w;
        p[12] = r3.x; p[13] = r3.y; p[14] = r3.z; p[15] = r3.w;
        v16bf a;
#pragma unroll
        for (int i = 0; i < 16; ++i) {
            p[i] = __expf(p[i] - mrow) * inv;
            a[i] = (bf16_t)p[i];
        }
        r0 = make_float4(p[0], p[1], p[2], p[3]);
        r1 = make_float4(p[4], p[5], p[6], p[7]);
        r2 = make_float4(p[8], p[9], p[10], p[11]);
        r3 = make_float4(p[12], p[13], p[14], p[15]);
        *(float4*)(arow + ks + sel)      = r0;
        *(float4*)(arow + ks + sel + 4)  = r1;
        *(float4*)(arow + ks + sel + 16) = r2;
        *(float4*)(arow + ks + sel + 20) = r3;
#pragma unroll
        for (int nt = 0; nt < 4; ++nt) {
            const bf16_t* vrow = vt + ((size_t)bh * DK + nt * 16 + lr) * SEQ + ks + sel;
            v16bf b = frag_from(vrow);
            c[nt] = wmma_bf16(a, b, c[nt]);
        }
    }

    const int b_ = bh >> 4;
    const int h  = bh & 15;
#pragma unroll
    for (int nt = 0; nt < 4; ++nt) {
#pragma unroll
        for (int v = 0; v < 8; ++v) {
            const int l = q0 + v + ((lane >> 4) << 3);
            const int d = nt * 16 + lr;
            ctxb[((size_t)(b_ * SEQ + l)) * D_MODEL + h * DK + d] = (bf16_t)c[nt][v];
        }
    }
}

// ---------------------------------------------------------------------------
// Output projection: out[m,n] = ctx[m,:] @ Wo[:,n] + bo[n], f32 out.
// Same LDS double-buffered B staging as gemm_qkv.
// ---------------------------------------------------------------------------
__global__ void __launch_bounds__(256) gemm_out_kernel(const bf16_t* __restrict__ ctxb,
                                                       const bf16_t* __restrict__ wT,
                                                       const float* __restrict__ bias,
                                                       float* __restrict__ out) {
    __shared__ bf16_t ldsB[2][BTILE_ELEMS];
    const int lane = threadIdx.x & 31;
    const int wave = threadIdx.x >> 5;
    const int m0   = blockIdx.x * 128 + wave * 16;
    const int n0   = blockIdx.y * 64;
    const int lr   = lane & 15;
    const int sel  = (lane >> 4) * 8;

    v8f c[4] = {};
    const bf16_t* arow = ctxb + (size_t)(m0 + lr) * D_MODEL;

    stage_b_tile(wT, n0, 0, ldsB[0]);
    stage_commit();
#pragma unroll 2
    for (int kt = 0; kt < D_MODEL / 32; ++kt) {
        const int cur = kt & 1;
        if (kt + 1 < D_MODEL / 32)
            stage_b_tile(wT, n0, (kt + 1) * 32, ldsB[cur ^ 1]);
        v16bf a = frag_from(arow + kt * 32 + sel);
#pragma unroll
        for (int nt = 0; nt < 4; ++nt) {
            v16bf b = frag_from(&ldsB[cur][(nt * 16 + lr) * BSTRIDE + sel]);
            c[nt] = wmma_bf16(a, b, c[nt]);
        }
        stage_commit();
    }

#pragma unroll
    for (int nt = 0; nt < 4; ++nt) {
        const int n   = n0 + nt * 16 + lr;
        const float bv = bias[n];
#pragma unroll
        for (int v = 0; v < 8; ++v) {
            const int m = m0 + v + ((lane >> 4) << 3);
            out[(size_t)m * D_MODEL + n] = c[nt][v] + bv;
        }
    }
}

// ---------------------------------------------------------------------------
// Host launcher
// ---------------------------------------------------------------------------
extern "C" void kernel_launch(void* const* d_in, const int* in_sizes, int n_in,
                              void* d_out, int out_size, void* d_ws, size_t ws_size,
                              hipStream_t stream) {
    (void)in_sizes; (void)n_in; (void)out_size; (void)ws_size;
    const float* x    = (const float*)d_in[0];  // [2,2048,1024]
    const float* Wqkv = (const float*)d_in[1];  // [1024,3072]
    const float* bqkv = (const float*)d_in[2];  // [3072]
    const float* Wo   = (const float*)d_in[3];  // [1024,1024]
    const float* bo   = (const float*)d_in[4];  // [1024]

    float* out  = (float*)d_out;                         // [2,2048,1024]
    float* attn = out + (size_t)MTOT * D_MODEL;          // [2,16,2048,2048]

    // workspace carve-up (bytes)
    char* ws = (char*)d_ws;
    bf16_t* xb    = (bf16_t*)(ws);                              // 8 MiB
    bf16_t* wqkvT = (bf16_t*)(ws + (8u << 20));                 // 6 MiB
    bf16_t* woT   = (bf16_t*)(ws + (14u << 20));                // 2 MiB
    bf16_t* qb    = (bf16_t*)(ws + (16u << 20));                // 8 MiB
    bf16_t* kb    = (bf16_t*)(ws + (24u << 20));                // 8 MiB
    bf16_t* vt    = (bf16_t*)(ws + (32u << 20));                // 8 MiB
    bf16_t* ctxb  = (bf16_t*)(ws + (40u << 20));                // 8 MiB
    float*  stats = (float*)(ws + (48u << 20));                 // 512 KiB

    // 1) precision / layout prep
    cvt_bf16_kernel<<<(MTOT * D_MODEL) / 256, 256, 0, stream>>>(x, xb, MTOT * D_MODEL);
    transpose_cvt_kernel<<<(D_MODEL * 3 * D_MODEL) / 256, 256, 0, stream>>>(Wqkv, wqkvT, D_MODEL, 3 * D_MODEL);
    transpose_cvt_kernel<<<(D_MODEL * D_MODEL) / 256, 256, 0, stream>>>(Wo, woT, D_MODEL, D_MODEL);

    // 2) QKV projection (M=4096, N=3072, K=1024)
    gemm_qkv_kernel<<<dim3(MTOT / 128, (3 * D_MODEL) / 64), 256, 0, stream>>>(
        xb, wqkvT, bqkv, qb, kb, vt);

    // 3) scores + online softmax stats (raw scaled scores -> attn region)
    attn_scores_kernel<<<dim3(BH, SEQ / 16 / 8), 256, 0, stream>>>(qb, kb, attn, stats);

    // 4) normalize attn in place + ctx = attn @ v
    attn_ctx_kernel<<<dim3(BH, SEQ / 16 / 8), 256, 0, stream>>>(vt, stats, attn, ctxb);

    // 5) output projection (M=4096, N=1024, K=1024)
    gemm_out_kernel<<<dim3(MTOT / 128, D_MODEL / 64), 256, 0, stream>>>(ctxb, woT, bo, out);
}
